// FlashMultiHeadAttention_78039555768488
// MI455X (gfx1250) — compile-verified, matrix operand order checked by reading, not run on hardware
//
#include <hip/hip_runtime.h>
#include <hip/hip_bf16.h>

typedef __attribute__((ext_vector_type(16))) _Float16 v16h;
typedef __attribute__((ext_vector_type(4)))  _Float16 v4h;
typedef __attribute__((ext_vector_type(8)))  float    v8f;
typedef __attribute__((ext_vector_type(4)))  unsigned int v4u;

#define HB 4      // batch
#define TT 2048   // sequence
#define DD 1024   // model dim
#define HH 16     // heads
#define DH 64     // head dim

union FragU { v16h h; v4u u[2]; };

// 16x32 f16 WMMA fragment load (A-layout; identical for B when source is
// N-major / K-contiguous). lane r<16: row r, halves 0..7 = K 8*hi.., halves
// 8..15 = K 16+8*hi.. (hi = lane>=16).  ld = row stride in halves.
__device__ __forceinline__ v16h load_frag(const _Float16* base, int ld, int lane) {
  const int r  = lane & 15;
  const int hi = lane >> 4;
  const _Float16* p = base + (size_t)r * ld + hi * 8;
  FragU f;
  f.u[0] = *(const v4u*)(p);
  f.u[1] = *(const v4u*)(p + 16);
  return f.h;
}

__device__ __forceinline__ v8f wmma16(v16h a, v16h b, v8f c) {
  return __builtin_amdgcn_wmma_f32_16x16x32_f16(false, a, false, b, (short)0, c,
                                                false, false);
}

// --- CDNA5 async global->LDS copy helpers (ASYNCcnt-tracked) ---------------
// Each active lane copies 16B per op: LDS[la + inst_off] = MEM[ga + inst_off].
__device__ __forceinline__ void async_copy_tiles(unsigned la, unsigned lb,
                                                 const _Float16* ga,
                                                 const _Float16* gb) {
  asm volatile(
      "global_load_async_to_lds_b128 %0, %2, off\n\t"
      "global_load_async_to_lds_b128 %0, %2, off offset:16\n\t"
      "global_load_async_to_lds_b128 %0, %2, off offset:32\n\t"
      "global_load_async_to_lds_b128 %0, %2, off offset:48\n\t"
      "global_load_async_to_lds_b128 %1, %3, off\n\t"
      "global_load_async_to_lds_b128 %1, %3, off offset:16"
      :
      : "v"(la), "v"(lb), "v"((unsigned long long)(size_t)ga),
        "v"((unsigned long long)(size_t)gb)
      : "memory");
}

__device__ __forceinline__ void wait_async_le6() {
#if __has_builtin(__builtin_amdgcn_s_wait_asynccnt)
  __builtin_amdgcn_s_wait_asynccnt(6);
#else
  asm volatile("s_wait_asynccnt 0x6" ::: "memory");
#endif
}
__device__ __forceinline__ void wait_async_le0() {
#if __has_builtin(__builtin_amdgcn_s_wait_asynccnt)
  __builtin_amdgcn_s_wait_asynccnt(0);
#else
  asm volatile("s_wait_asynccnt 0x0" ::: "memory");
#endif
}

// ---------------------------------------------------------------------------
// Shared GEMM mainloop: 256x128 block tile, K-step 32, double-buffered LDS
// filled by async global->LDS DMA. 256 threads = 8 waves (4x2); each wave
// owns a 64x64 sub-tile = 4x4 frags -> 16 WMMAs per 8 fragment loads.
// ---------------------------------------------------------------------------
__device__ __forceinline__ void gemm_mainloop(
    const _Float16* __restrict__ A, const _Float16* __restrict__ BT,
    int m0, int n0, _Float16* ldsA, _Float16* ldsB, int tid, v8f acc[4][4]) {
  const int lane = tid & 31;
  const int wave = tid >> 5;
  const int wm   = (wave & 3) * 64;   // 0,64,128,192
  const int wn   = (wave >> 2) * 64;  // 0,64
  const int brow = tid >> 1;          // 0..127 (B tile rows)
  const int bch  = (tid & 1) * 16;    // 0 or 16 halves

  const _Float16* Ar = A  + (size_t)(m0 + tid)  * DD;        // one row/thread
  const _Float16* Br = BT + (size_t)(n0 + brow) * DD + bch;

  // Per-lane LDS byte addresses (low 32 bits of the generic pointer are the
  // LDS aperture offset).
  const unsigned laBase = (unsigned)(size_t)(ldsA) + (unsigned)tid * 64u;
  const unsigned lbBase =
      (unsigned)(size_t)(ldsB) + (unsigned)brow * 64u + (unsigned)bch * 2u;

  // Prologue: fill buffer 0.
  async_copy_tiles(laBase, lbBase, Ar, Br);

  for (int k0 = 0; k0 < DD; k0 += 32) {
    const int cur = (k0 >> 5) & 1;
    if (k0 + 32 < DD) {
      // Kick off next tile into the other buffer, then ensure the current
      // tile's 6 async ops (issued last iteration) have landed.
      async_copy_tiles(laBase + (unsigned)(1 - cur) * 16384u,
                       lbBase + (unsigned)(1 - cur) * 8192u,
                       Ar + k0 + 32, Br + k0 + 32);
      wait_async_le6();
    } else {
      wait_async_le0();
    }
    __syncthreads();   // current tile visible to all waves

    const _Float16* tA = ldsA + (size_t)cur * (256 * 32);
    const _Float16* tB = ldsB + (size_t)cur * (128 * 32);
    const v16h a0 = load_frag(tA + (size_t)(wm)      * 32, 32, lane);
    const v16h a1 = load_frag(tA + (size_t)(wm + 16) * 32, 32, lane);
    const v16h a2 = load_frag(tA + (size_t)(wm + 32) * 32, 32, lane);
    const v16h a3 = load_frag(tA + (size_t)(wm + 48) * 32, 32, lane);
    const v16h b0 = load_frag(tB + (size_t)(wn)      * 32, 32, lane);
    const v16h b1 = load_frag(tB + (size_t)(wn + 16) * 32, 32, lane);
    const v16h b2 = load_frag(tB + (size_t)(wn + 32) * 32, 32, lane);
    const v16h b3 = load_frag(tB + (size_t)(wn + 48) * 32, 32, lane);
    acc[0][0] = wmma16(a0, b0, acc[0][0]);
    acc[1][0] = wmma16(a1, b0, acc[1][0]);
    acc[2][0] = wmma16(a2, b0, acc[2][0]);
    acc[3][0] = wmma16(a3, b0, acc[3][0]);
    acc[0][1] = wmma16(a0, b1, acc[0][1]);
    acc[1][1] = wmma16(a1, b1, acc[1][1]);
    acc[2][1] = wmma16(a2, b1, acc[2][1]);
    acc[3][1] = wmma16(a3, b1, acc[3][1]);
    acc[0][2] = wmma16(a0, b2, acc[0][2]);
    acc[1][2] = wmma16(a1, b2, acc[1][2]);
    acc[2][2] = wmma16(a2, b2, acc[2][2]);
    acc[3][2] = wmma16(a3, b2, acc[3][2]);
    acc[0][3] = wmma16(a0, b3, acc[0][3]);
    acc[1][3] = wmma16(a1, b3, acc[1][3]);
    acc[2][3] = wmma16(a2, b3, acc[2][3]);
    acc[3][3] = wmma16(a3, b3, acc[3][3]);
    __syncthreads();   // readers done before next iteration overwrites `cur`
  }
}

// ---------------------------------------------------------------------------
// Kernel 1: f32 -> f16 convert (vectorized)
// ---------------------------------------------------------------------------
__global__ __launch_bounds__(256) void cvt_kernel(const float* __restrict__ x,
                                                  _Float16* __restrict__ xh,
                                                  int n4) {
  const int i = blockIdx.x * 256 + threadIdx.x;
  if (i < n4) {
    const float4 f = ((const float4*)x)[i];
    v4h hv;
    hv.x = (_Float16)f.x; hv.y = (_Float16)f.y;
    hv.z = (_Float16)f.z; hv.w = (_Float16)f.w;
    ((v4h*)xh)[i] = hv;
  }
}

// ---------------------------------------------------------------------------
// Kernel 2: transposed convert: WT[n*K + k] = (f16) W[k*N + n]
// ---------------------------------------------------------------------------
__global__ __launch_bounds__(256) void transpose_cvt(const float* __restrict__ W,
                                                     _Float16* __restrict__ WT,
                                                     int K, int N) {
  __shared__ float tile[32][33];
  const int n0 = blockIdx.x * 32, k0 = blockIdx.y * 32;
  const int tx = threadIdx.x, ty = threadIdx.y;  // block (32, 8)
  for (int i = ty; i < 32; i += 8)
    tile[i][tx] = W[(size_t)(k0 + i) * N + n0 + tx];
  __syncthreads();
  for (int i = ty; i < 32; i += 8)
    WT[(size_t)(n0 + i) * K + k0 + tx] = (_Float16)tile[tx][i];
}

// ---------------------------------------------------------------------------
// Kernel 3: QKV GEMM + bias + RoPE, scatter into Q[b,h,t,d], K[b,h,t,d],
// V^T[b,h,d,t] (all f16).  grid = (3072/128, 8192/256)
// ---------------------------------------------------------------------------
__global__ __launch_bounds__(256) void qkv_rope_kernel(
    const _Float16* __restrict__ Xh, const _Float16* __restrict__ WT,
    const float* __restrict__ bqkv, const float* __restrict__ cosT,
    const float* __restrict__ sinT, _Float16* __restrict__ Qb,
    _Float16* __restrict__ Kb, _Float16* __restrict__ VT) {
  __shared__ alignas(16) _Float16 ldsA[2 * 256 * 32];
  __shared__ alignas(16) _Float16 ldsB[2 * 128 * 32];
  const int tid  = threadIdx.x;
  const int lane = tid & 31;
  const int hi   = lane >> 4;
  const int cn   = lane & 15;
  const int wave = tid >> 5;
  const int wm   = (wave & 3) * 64;
  const int wn   = (wave >> 2) * 64;
  const int m0   = blockIdx.y * 256;
  const int n0   = blockIdx.x * 128;

  v8f acc[4][4] = {};
  gemm_mainloop(Xh, WT, m0, n0, ldsA, ldsB, tid, acc);

  // Wave's 64-col span == exactly one head (n0 mult of 128, wn in {0,64}),
  // so rotate-half pairs (d, d+32) live in acc[i][jj] / acc[i][jj+2].
#pragma unroll
  for (int i = 0; i < 4; ++i) {
#pragma unroll
    for (int jj = 0; jj < 2; ++jj) {
      const int gn_lo = n0 + wn + 16 * jj + cn;
      const int gn_hi = gn_lo + 32;
      const int sec   = gn_lo >> 10;   // 0=Q 1=K 2=V
      const int nn    = gn_lo & 1023;
      const int h     = nn >> 6;
      const int dlo   = nn & 63;       // < 32
      const int dhi   = dlo + 32;
      const float blo = bqkv[gn_lo];
      const float bhi = bqkv[gn_hi];
#pragma unroll
      for (int v = 0; v < 8; ++v) {
        const int gm = m0 + wm + 16 * i + v + hi * 8;
        const int b  = gm >> 11;
        const int t  = gm & (TT - 1);
        const float xlo = acc[i][jj][v]     + blo;
        const float xhi = acc[i][jj + 2][v] + bhi;
        if (sec == 2) {
          VT[((size_t)((b * HH + h) * DH + dlo)) * TT + t] = (_Float16)xlo;
          VT[((size_t)((b * HH + h) * DH + dhi)) * TT + t] = (_Float16)xhi;
        } else {
          const float cl = cosT[t * DH + dlo], sl = sinT[t * DH + dlo];
          const float ch = cosT[t * DH + dhi], sh = sinT[t * DH + dhi];
          _Float16* dst = (sec == 0) ? Qb : Kb;
          dst[((size_t)((b * HH + h) * TT + t)) * DH + dlo] =
              (_Float16)(xlo * cl - xhi * sl);
          dst[((size_t)((b * HH + h) * TT + t)) * DH + dhi] =
              (_Float16)(xhi * ch + xlo * sh);
        }
      }
    }
  }
}

// ---------------------------------------------------------------------------
// Kernel 4: flash attention. grid = (T/128, B*H); 8 waves/block, each wave
// owns 16 query rows and streams 32 keys per iteration.
// ---------------------------------------------------------------------------
__global__ __launch_bounds__(256) void attn_kernel(
    const _Float16* __restrict__ Q, const _Float16* __restrict__ K,
    const _Float16* __restrict__ VT, _Float16* __restrict__ O) {
  __shared__ alignas(16) _Float16 Ptile[8][16 * 32];
  const int tid  = threadIdx.x;
  const int lane = tid & 31;
  const int hi   = lane >> 4;
  const int cn   = lane & 15;
  const int wave = tid >> 5;
  const int bh   = blockIdx.y;
  const int b    = bh >> 4;
  const int h    = bh & 15;
  const int tq0  = blockIdx.x * 128 + wave * 16;

  const _Float16* Qp = Q  + ((size_t)bh * TT + tq0) * DH;
  const _Float16* Kp = K  + (size_t)bh * TT * DH;
  const _Float16* Vp = VT + (size_t)bh * DH * TT;   // [DH][TT]
  _Float16* myP = &Ptile[wave][0];

  const v16h q0 = load_frag(Qp,      DH, lane);     // d 0..31
  const v16h q1 = load_frag(Qp + 32, DH, lane);     // d 32..63

  float mrow[8], lrow[8];
#pragma unroll
  for (int v = 0; v < 8; ++v) { mrow[v] = -3.0e38f; lrow[v] = 0.f; }
  v8f o[4] = {};

  for (int kb = 0; kb < TT; kb += 32) {
    v8f s0 = {}, s1 = {};
    const v16h k00 = load_frag(Kp + (size_t)kb * DH,             DH, lane);
    const v16h k01 = load_frag(Kp + (size_t)kb * DH + 32,        DH, lane);
    const v16h k10 = load_frag(Kp + (size_t)(kb + 16) * DH,      DH, lane);
    const v16h k11 = load_frag(Kp + (size_t)(kb + 16) * DH + 32, DH, lane);
    if (kb + 32 < TT) {
      __builtin_prefetch(Kp + (size_t)(kb + 32) * DH, 0, 1);
      __builtin_prefetch(Vp + (size_t)cn * TT + kb + 32, 0, 1);
    }
    s0 = wmma16(q0, k00, s0);
    s0 = wmma16(q1, k01, s0);
    s1 = wmma16(q0, k10, s1);
    s1 = wmma16(q1, k11, s1);

    float alpha[8];
#pragma unroll
    for (int v = 0; v < 8; ++v) {
      const float a = (float)s0[v] * 0.125f;   // 1/sqrt(64)
      const float c = (float)s1[v] * 0.125f;
      float mx = fmaxf(a, c);                  // row = 16-lane group
#pragma unroll
      for (int off = 8; off >= 1; off >>= 1)
        mx = fmaxf(mx, __shfl_xor(mx, off, 16));
      const float mnew = fmaxf(mrow[v], mx);
      const float al   = __expf(mrow[v] - mnew);
      const float p0   = __expf(a - mnew);
      const float p1   = __expf(c - mnew);
      float sum = p0 + p1;
#pragma unroll
      for (int off = 8; off >= 1; off >>= 1)
        sum += __shfl_xor(sum, off, 16);
      lrow[v]  = lrow[v] * al + sum;
      mrow[v]  = mnew;
      alpha[v] = al;
      const int mloc = v + hi * 8;             // D-layout -> A-layout via LDS
      myP[mloc * 32 + cn]      = (_Float16)p0;
      myP[mloc * 32 + 16 + cn] = (_Float16)p1;
    }

    const v16h pf  = load_frag(myP, 32, lane); // same-wave DS ops in order
    const v16h vf0 = load_frag(Vp + (size_t)0  * TT + kb, TT, lane);
    const v16h vf1 = load_frag(Vp + (size_t)16 * TT + kb, TT, lane);
    const v16h vf2 = load_frag(Vp + (size_t)32 * TT + kb, TT, lane);
    const v16h vf3 = load_frag(Vp + (size_t)48 * TT + kb, TT, lane);
#pragma unroll
    for (int f = 0; f < 4; ++f)
#pragma unroll
      for (int v = 0; v < 8; ++v) o[f][v] *= alpha[v];
    o[0] = wmma16(pf, vf0, o[0]);
    o[1] = wmma16(pf, vf1, o[1]);
    o[2] = wmma16(pf, vf2, o[2]);
    o[3] = wmma16(pf, vf3, o[3]);
  }

  float inv[8];
#pragma unroll
  for (int v = 0; v < 8; ++v) inv[v] = 1.0f / lrow[v];
#pragma unroll
  for (int f = 0; f < 4; ++f) {
#pragma unroll
    for (int v = 0; v < 8; ++v) {
      const int tq = tq0 + v + hi * 8;
      const int d  = 16 * f + cn;
      O[((size_t)(b * TT + tq)) * DD + h * DH + d] = (_Float16)(o[f][v] * inv[v]);
    }
  }
}

// ---------------------------------------------------------------------------
// Kernel 5: output projection, fp32 result + bias. grid = (1024/128, 8192/256)
// ---------------------------------------------------------------------------
__global__ __launch_bounds__(256) void proj_kernel(
    const _Float16* __restrict__ Ah, const _Float16* __restrict__ WT,
    const float* __restrict__ bproj, float* __restrict__ out) {
  __shared__ alignas(16) _Float16 ldsA[2 * 256 * 32];
  __shared__ alignas(16) _Float16 ldsB[2 * 128 * 32];
  const int tid  = threadIdx.x;
  const int lane = tid & 31;
  const int hi   = lane >> 4;
  const int cn   = lane & 15;
  const int wave = tid >> 5;
  const int wm   = (wave & 3) * 64;
  const int wn   = (wave >> 2) * 64;
  const int m0   = blockIdx.y * 256;
  const int n0   = blockIdx.x * 128;

  v8f acc[4][4] = {};
  gemm_mainloop(Ah, WT, m0, n0, ldsA, ldsB, tid, acc);

#pragma unroll
  for (int i = 0; i < 4; ++i) {
#pragma unroll
    for (int j = 0; j < 4; ++j) {
      const int gn = n0 + wn + 16 * j + cn;
      const float bv = bproj[gn];
#pragma unroll
      for (int v = 0; v < 8; ++v) {
        const int gm = m0 + wm + 16 * i + v + hi * 8;
        out[(size_t)gm * DD + gn] = acc[i][j][v] + bv;
      }
    }
  }
}

// ---------------------------------------------------------------------------
extern "C" void kernel_launch(void* const* d_in, const int* in_sizes, int n_in,
                              void* d_out, int out_size, void* d_ws,
                              size_t ws_size, hipStream_t stream) {
  const float* x     = (const float*)d_in[0];
  const float* Wqkv  = (const float*)d_in[1];
  const float* bqkv  = (const float*)d_in[2];
  const float* Wproj = (const float*)d_in[3];
  const float* bproj = (const float*)d_in[4];
  const float* cosT  = (const float*)d_in[5];
  const float* sinT  = (const float*)d_in[6];
  float* out = (float*)d_out;

  char* ws = (char*)d_ws;
  // byte offsets (total 72 MB)
  _Float16* xh     = (_Float16*)(ws);              // 16 MB, reused as attn out
  _Float16* wqkvT  = (_Float16*)(ws + 16777216);   //  6 MB
  _Float16* wprojT = (_Float16*)(ws + 23068672);   //  2 MB
  _Float16* Qb     = (_Float16*)(ws + 25165824);   // 16 MB
  _Float16* Kb     = (_Float16*)(ws + 41943040);   // 16 MB
  _Float16* VT     = (_Float16*)(ws + 58720256);   // 16 MB
  _Float16* attnh  = xh;   // x no longer needed after QKV GEMM

  // 1) precision conversion / weight transpose
  cvt_kernel<<<8192, 256, 0, stream>>>(x, xh, (HB * TT * DD) / 4);
  transpose_cvt<<<dim3(96, 32), dim3(32, 8), 0, stream>>>(Wqkv, wqkvT, DD, 3 * DD);
  transpose_cvt<<<dim3(32, 32), dim3(32, 8), 0, stream>>>(Wproj, wprojT, DD, DD);
  // 2) QKV GEMM + RoPE  (M=8192 in 256-row tiles, N=3072 in 128-col tiles)
  qkv_rope_kernel<<<dim3(24, 32), 256, 0, stream>>>(xh, wqkvT, bqkv, cosT, sinT,
                                                    Qb, Kb, VT);
  // 3) flash attention  (grid: T/128 x B*H)
  attn_kernel<<<dim3(TT / 128, HB * HH), 256, 0, stream>>>(Qb, Kb, VT, attnh);
  // 4) output projection (M=8192, N=1024)
  proj_kernel<<<dim3(8, 32), 256, 0, stream>>>(attnh, wprojT, bproj, out);
}
